// Atentie_17093969838083
// MI455X (gfx1250) — matrix operand order repaired and checked by reading, not compiled, not run on hardware
//
#include <hip/hip_runtime.h>
#include <hip/hip_bf16.h>
#include <math.h>

typedef __attribute__((ext_vector_type(16))) _Float16 v16h;
typedef __attribute__((ext_vector_type(8)))  _Float16 v8h;
typedef __attribute__((ext_vector_type(8)))  float    v8f;
typedef __attribute__((ext_vector_type(4)))  int      v4i;

#define BATCH 4
#define SEQ   2048
#define DIM   1024
#define ROWS  (BATCH * SEQ)        // 8192
#define SCALE 0.03125f             // 1/sqrt(1024)

#if __has_builtin(__builtin_amdgcn_global_load_async_to_lds_b128) && \
    __has_builtin(__builtin_amdgcn_s_wait_asynccnt)
#define HAVE_ASYNC_LDS 1
typedef __attribute__((address_space(1))) v4i g_v4i;   // global-AS int4 (param type per hipcc diag)
typedef __attribute__((address_space(3))) v4i l_v4i;   // LDS-AS int4
#else
#define HAVE_ASYNC_LDS 0
#endif

// ---------------- WMMA fragment loaders (16x16x32 f16; contiguous K, same order for A and B) ----------------
// A: 16x32 (MxK). lane<16: row=lane, K = {k0..k0+15}
//                 lane>=16: row=lane-16, K = {k0+16..k0+31}
__device__ inline v16h load_a_frag_f16(const _Float16* base, int ld, int m0, int k0, int lane) {
    int r  = lane & 15;
    int hi = (lane >> 4) << 4;            // 0 or 16
    const _Float16* p = base + (size_t)(m0 + r) * ld + k0 + hi;
    v16h a;
    *((v8h*)&a)     = *(const v8h*)p;          // K = k0+hi+0..7
    *((v8h*)&a + 1) = *(const v8h*)(p + 8);    // K = k0+hi+8..15
    return a;
}

// B: 32x16 (KxN) with B[k][n] = R[n0+n][k0+k] for row-major R (ld = row stride).
// lane holds column n = lane&15; 16 contiguous K halves, +16 for upper lanes.
__device__ inline v16h load_b_frag_rows_f16(const _Float16* base, int ld, int n0, int k0, int lane) {
    int n  = lane & 15;
    int hi = (lane >> 4) << 4;            // 0 or 16
    const _Float16* p = base + (size_t)(n0 + n) * ld + k0 + hi;
    v16h b;
    *((v8h*)&b)     = *(const v8h*)p;
    *((v8h*)&b + 1) = *(const v8h*)(p + 8);
    return b;
}

// ---------------- Kernel 0: one-shot fp32 -> f16 conversion ----------------
__global__ __launch_bounds__(256) void cvt_kernel(const float* __restrict__ src,
                                                  _Float16* __restrict__ dst, int n)
{
    int i = (blockIdx.x * 256 + threadIdx.x) * 8;
    if (i >= n) return;
    v8f x = *(const v8f*)(src + i);
    v8h h;
#pragma unroll
    for (int j = 0; j < 8; ++j) h[j] = (_Float16)x[j];
    *(v8h*)(dst + i) = h;
}

// ---------------- Kernel 1: QKV projection GEMM (f16 in, f16 out) ----------------
// Xh[8192,1024] @ Wh^T where Wh = [Wq;Wk;Wv] is [3072,1024] f16.
// Each wave computes a 16x64 strip: A-fragment reused across 4 independent
// accumulators (no WMMA RAW chain, 4x less A traffic).
// n in [0,1024)->Q, [1024,2048)->K (row-major), [2048,3072)->V stored
// transposed as Vt[b, DIM, SEQ] for the later P@V GEMM.
__global__ __launch_bounds__(256) void qkv_gemm(
    const _Float16* __restrict__ Xh, const _Float16* __restrict__ Wh,
    const float* __restrict__ bq, const float* __restrict__ bk, const float* __restrict__ bv,
    _Float16* __restrict__ Q, _Float16* __restrict__ K, _Float16* __restrict__ Vt)
{
    const int lane = threadIdx.x & 31;
    const int gw   = blockIdx.x * 8 + (threadIdx.x >> 5);

    const int NT = (3 * DIM) / 64;            // 48 strips of width 64
    const int nt = gw % NT;
    const int mt = gw / NT;
    const int m0 = mt << 4;
    const int n0 = nt << 6;                   // global n over concatenated [Q|K|V]

    v8f acc[4];
#pragma unroll
    for (int t = 0; t < 4; ++t) acc[t] = (v8f){};

#pragma unroll 2
    for (int k0 = 0; k0 < DIM; k0 += 32) {
        v16h a = load_a_frag_f16(Xh, DIM, m0, k0, lane);
#pragma unroll
        for (int t = 0; t < 4; ++t) {
            v16h b = load_b_frag_rows_f16(Wh, DIM, n0 + t * 16, k0, lane);
            acc[t] = __builtin_amdgcn_wmma_f32_16x16x32_f16(false, a, false, b,
                                                            (short)0, acc[t], false, false);
        }
    }

    const int mat   = n0 >> 10;               // 0=Q 1=K 2=V (64 | 1024 so uniform)
    const int nloc0 = n0 & (DIM - 1);
    const float* bias = (mat == 0) ? bq : (mat == 1) ? bk : bv;
    const int colN  = lane & 15;
    const int rbase = m0 + ((lane >> 4) << 3);

#pragma unroll
    for (int t = 0; t < 4; ++t) {
        const int nl = nloc0 + t * 16 + colN;
        const float badd = bias[nl];
        if (mat < 2) {
            _Float16* dst = (mat == 0) ? Q : K;
#pragma unroll
            for (int v = 0; v < 8; ++v)
                dst[(size_t)(rbase + v) * DIM + nl] = (_Float16)(acc[t][v] + badd);
        } else {
#pragma unroll
            for (int v = 0; v < 8; ++v) {
                int s  = rbase + v;
                Vt[((size_t)(s >> 11) * DIM + nl) * SEQ + (s & (SEQ - 1))] =
                    (_Float16)(acc[t][v] + badd);
            }
        }
    }
}

// ---------------- Kernel 2: streaming (flash-style) attention ----------------
// One block per (batch, 16-query tile). 8 waves; wave w owns:
//   - score keys [j0 + w*16, +16) each 128-key chunk (full-K QK^T tile)
//   - output D-slice [w*128, +128) (8 accumulator tiles)
__global__ __launch_bounds__(256) void attn_kernel(
    const _Float16* __restrict__ Q,
    const _Float16* __restrict__ K,
    const _Float16* __restrict__ Vt,
    float* __restrict__ Out)
{
    __shared__ _Float16 Qs[16 * DIM];          // 32 KB
    __shared__ float    Sc[16 * 128];          // 8 KB  (scores chunk)
    __shared__ _Float16 Pls[16 * 128];         // 4 KB  (softmax probs, f16)
    __shared__ float    red[16 * 16];          // 1 KB  (max/sum partials)
    __shared__ float    mRow[16], lRow[16], aRow[16];

    const int tid  = threadIdx.x;
    const int lane = tid & 31;
    const int wave = tid >> 5;
    const int b    = blockIdx.x >> 7;
    const int q0   = (blockIdx.x & 127) << 4;

    // ---- Stage Q tile (16 consecutive rows -> contiguous 16K halves) ----
    const _Float16* Qb = Q + ((size_t)b * SEQ + q0) * DIM;
#if HAVE_ASYNC_LDS
    for (int i = tid * 8; i < 16 * DIM; i += 256 * 8)
        __builtin_amdgcn_global_load_async_to_lds_b128(
            (g_v4i*)(Qb + i), (l_v4i*)(Qs + i), 0, 0);
    __builtin_amdgcn_s_wait_asynccnt(0);
#else
    for (int i = tid * 8; i < 16 * DIM; i += 256 * 8)
        *(v8h*)(Qs + i) = *(const v8h*)(Qb + i);
#endif

    if (tid < 16) { mRow[tid] = -INFINITY; lRow[tid] = 0.0f; }

    v8f acc[8];
#pragma unroll
    for (int t = 0; t < 8; ++t) acc[t] = (v8f){};

    const int colN  = lane & 15;
    const int rbase = (lane >> 4) << 3;
    const int srow  = tid >> 4;                 // softmax row handled by this thread
    const int scol  = (tid & 15) << 3;          // 8-col slice

    __syncthreads();

    for (int j0 = 0; j0 < SEQ; j0 += 128) {
        const _Float16* Kb = K + ((size_t)b * SEQ + j0 + wave * 16) * DIM;

        // Prefetch next chunk's K and Vt rows (1 line per lane, hides first touch)
        if (j0 + 128 < SEQ) {
            const _Float16* Kn = Kb + (size_t)128 * DIM;
            __builtin_prefetch(Kn + (size_t)(lane & 15) * DIM + ((lane >> 4) << 9), 0, 3);
            const _Float16* Vn = Vt + ((size_t)b * DIM + wave * 128 + lane * 4) * SEQ + j0 + 128;
            __builtin_prefetch(Vn, 0, 3);
        }

        // ---- scores: two independent WMMA chains over K=1024 ----
        v8f s0 = {}, s1 = {};
#pragma unroll 2
        for (int k0 = 0; k0 < DIM; k0 += 64) {
            v16h a0 = load_a_frag_f16(Qs, DIM, 0, k0, lane);
            v16h b0 = load_b_frag_rows_f16(Kb, DIM, 0, k0, lane);
            s0 = __builtin_amdgcn_wmma_f32_16x16x32_f16(false, a0, false, b0,
                                                        (short)0, s0, false, false);
            v16h a1 = load_a_frag_f16(Qs, DIM, 0, k0 + 32, lane);
            v16h b1 = load_b_frag_rows_f16(Kb, DIM, 0, k0 + 32, lane);
            s1 = __builtin_amdgcn_wmma_f32_16x16x32_f16(false, a1, false, b1,
                                                        (short)0, s1, false, false);
        }
#pragma unroll
        for (int v = 0; v < 8; ++v)
            Sc[(rbase + v) * 128 + wave * 16 + colN] = (s0[v] + s1[v]) * SCALE;

        __syncthreads();

        // ---- online softmax, parallel across all 256 threads ----
        {   // partial row max (8 cols per thread)
            const float* sr = Sc + srow * 128 + scol;
            float pm = sr[0];
#pragma unroll
            for (int j = 1; j < 8; ++j) pm = fmaxf(pm, sr[j]);
            red[srow * 16 + (tid & 15)] = pm;
        }
        __syncthreads();
        if (tid < 16) {
            float cm = mRow[tid];
#pragma unroll
            for (int i = 0; i < 16; ++i) cm = fmaxf(cm, red[tid * 16 + i]);
            aRow[tid] = __expf(mRow[tid] - cm);   // 0 on first chunk
            mRow[tid] = cm;
        }
        __syncthreads();
        {   // exp + probs + partial sums
            const float cm = mRow[srow];
            const float* sr = Sc + srow * 128 + scol;
            _Float16* pr = Pls + srow * 128 + scol;
            float psum = 0.0f;
#pragma unroll
            for (int j = 0; j < 8; ++j) {
                float p = __expf(sr[j] - cm);
                pr[j] = (_Float16)p;
                psum += p;
            }
            red[srow * 16 + (tid & 15)] = psum;
        }
        __syncthreads();
        if (tid < 16) {
            float s = 0.0f;
#pragma unroll
            for (int i = 0; i < 16; ++i) s += red[tid * 16 + i];
            lRow[tid] = lRow[tid] * aRow[tid] + s;
        }
        __syncthreads();

        // ---- rescale O accumulators ----
        float av[8];
#pragma unroll
        for (int v = 0; v < 8; ++v) av[v] = aRow[rbase + v];
#pragma unroll
        for (int t = 0; t < 8; ++t)
#pragma unroll
            for (int v = 0; v < 8; ++v) acc[t][v] *= av[v];

        // ---- O_slice += P @ V_chunk (K = 128, via Vt rows) ----
        const _Float16* Vb = Vt + ((size_t)b * DIM + wave * 128) * SEQ + j0;
#pragma unroll
        for (int kf = 0; kf < 4; ++kf) {
            v16h aP = load_a_frag_f16(Pls, 128, 0, kf * 32, lane);
#pragma unroll
            for (int t = 0; t < 8; ++t) {
                v16h bV = load_b_frag_rows_f16(Vb + kf * 32, SEQ, t * 16, 0, lane);
                acc[t] = __builtin_amdgcn_wmma_f32_16x16x32_f16(false, aP, false, bV,
                                                                (short)0, acc[t], false, false);
            }
        }
        // next iteration's Pls write is ordered by its first __syncthreads
    }

    // ---- finalize: divide by row sums, store fp32 ----
    float inv[8];
#pragma unroll
    for (int v = 0; v < 8; ++v) inv[v] = 1.0f / lRow[rbase + v];
#pragma unroll
    for (int t = 0; t < 8; ++t) {
#pragma unroll
        for (int v = 0; v < 8; ++v) {
            Out[((size_t)b * SEQ + q0 + rbase + v) * DIM + wave * 128 + t * 16 + colN] =
                acc[t][v] * inv[v];
        }
    }
}

// ---------------- host launcher ----------------
extern "C" void kernel_launch(void* const* d_in, const int* in_sizes, int n_in,
                              void* d_out, int out_size, void* d_ws, size_t ws_size,
                              hipStream_t stream)
{
    const float* X  = (const float*)d_in[0];
    const float* Wq = (const float*)d_in[1];
    const float* bq = (const float*)d_in[2];
    const float* Wk = (const float*)d_in[3];
    const float* bk = (const float*)d_in[4];
    const float* Wv = (const float*)d_in[5];
    const float* bv = (const float*)d_in[6];
    float* out = (float*)d_out;

    const size_t NX = (size_t)ROWS * DIM;      // 8M
    const size_t NW = (size_t)DIM * DIM;       // 1M per weight

    _Float16* Xh  = (_Float16*)d_ws;
    _Float16* Wh  = Xh + NX;                   // [3072,1024] = [Wq;Wk;Wv]
    _Float16* Qh  = Wh + 3 * NW;
    _Float16* Kh  = Qh + NX;
    _Float16* Vth = Kh + NX;                   // stored [B, DIM, SEQ]

    // One-shot f32->f16 conversions
    cvt_kernel<<<(int)(NX / 8 / 256), 256, 0, stream>>>(X,  Xh, (int)NX);
    cvt_kernel<<<(int)(NW / 8 / 256), 256, 0, stream>>>(Wq, Wh + 0 * NW, (int)NW);
    cvt_kernel<<<(int)(NW / 8 / 256), 256, 0, stream>>>(Wk, Wh + 1 * NW, (int)NW);
    cvt_kernel<<<(int)(NW / 8 / 256), 256, 0, stream>>>(Wv, Wh + 2 * NW, (int)NW);

    // QKV GEMM: 512 m-tiles x 48 n-strips, 8 waves per block
    const int total_waves = (ROWS / 16) * ((3 * DIM) / 64);
    qkv_gemm<<<total_waves / 8, 256, 0, stream>>>(Xh, Wh, bq, bk, bv, Qh, Kh, Vth);

    // Attention: one block per (batch, 16-query tile)
    attn_kernel<<<BATCH * (SEQ / 16), 256, 0, stream>>>(Qh, Kh, Vth, out);
}
